// Model_71665824301542
// MI455X (gfx1250) — matrix operand order repaired and checked by reading, not compiled, not run on hardware
//
#include <hip/hip_runtime.h>
#include <hip/hip_bf16.h>
#include <math.h>

// ---------------- problem constants ----------------
#define NW    2048          // num words
#define HID   768           // hidden
#define NC    8192          // candidates
#define MAXW  30            // max span width
#define FEAT  20
#define FF    1000          // ffnn width
#define SD    2324          // span dim = 3*HID + FEAT
#define KP1   2336          // SD padded to mult of 32
#define NP    1024          // FF padded to mult of 16 (and 32 for K of gemm2)
#define TOPN  819

#define MT    (NC/16)       // 512 M tiles
#define NTI   (NP/16)       // 64  N tiles
#define KT1   (KP1/32)      // 73
#define KT2   (NP/32)       // 32

// output layout (floats)
#define O_SCORES 0
#define O_TIDX   (NC)
#define O_TSTART (NC + TOPN)
#define O_TEND   (NC + 2*TOPN)
#define O_TEMB   (NC + 3*TOPN)
#define O_TSCORE (NC + 3*TOPN + TOPN*SD)

typedef __attribute__((ext_vector_type(16))) __bf16 v16bf;
typedef __attribute__((ext_vector_type(8)))  __bf16 v8bf;
typedef __attribute__((ext_vector_type(8)))  float  v8f;
typedef __attribute__((ext_vector_type(4)))  int    v4i;

#ifndef __has_builtin
#define __has_builtin(x) 0
#endif
#if __has_builtin(__builtin_amdgcn_global_load_async_to_lds_b128)
#define HAVE_ASYNC 1
#else
#define HAVE_ASYNC 0
#endif
#if __has_builtin(__builtin_amdgcn_s_wait_asynccnt)
#define HAVE_WAIT_ASYNC 1
#else
#define HAVE_WAIT_ASYNC 0
#endif

static __device__ __forceinline__ void async_wait0() {
#if HAVE_WAIT_ASYNC
  __builtin_amdgcn_s_wait_asynccnt(0);
#elif HAVE_ASYNC
  asm volatile("s_wait_asynccnt 0x0" ::: "memory");
#endif
}

static __device__ __forceinline__ v16bf load_afrag(const __bf16* p) {
  v8bf lo = *(const v8bf*)p;
  v8bf hi = *(const v8bf*)(p + 16);
  return __builtin_shufflevector(lo, hi, 0, 1, 2, 3, 4, 5, 6, 7,
                                 8, 9, 10, 11, 12, 13, 14, 15);
}

// ---------------- head scores: mention_doc @ head_w + head_b ----------------
__global__ void k_head(const float* __restrict__ md, const float* __restrict__ hw,
                       const float* __restrict__ hb, float* __restrict__ out) {
  __shared__ float red[256];
  int r = blockIdx.x;
  float p = 0.f;
  for (int h = threadIdx.x; h < HID; h += 256) p += md[r * HID + h] * hw[h];
  red[threadIdx.x] = p; __syncthreads();
  for (int s = 128; s > 0; s >>= 1) {
    if (threadIdx.x < s) red[threadIdx.x] += red[threadIdx.x + s];
    __syncthreads();
  }
  if (threadIdx.x == 0) out[r] = red[0] + hb[0];
}

// ---------------- width prior FFNN (tiny): one block per width ----------------
__global__ void k_width(const float* __restrict__ wpe,
                        const float* __restrict__ w1, const float* __restrict__ b1,
                        const float* __restrict__ w2, const float* __restrict__ b2,
                        const float* __restrict__ w3, const float* __restrict__ b3,
                        float* __restrict__ out) {
  __shared__ float h1[FF];
  __shared__ float h2[FF];
  __shared__ float red[256];
  int r = blockIdx.x;
  float x[FEAT];
  for (int k = 0; k < FEAT; ++k) x[k] = wpe[r * FEAT + k];
  for (int o = threadIdx.x; o < FF; o += 256) {
    float a = b1[o];
    for (int k = 0; k < FEAT; ++k) a += x[k] * w1[k * FF + o];
    h1[o] = a > 0.f ? a : 0.f;
  }
  __syncthreads();
  for (int o = threadIdx.x; o < FF; o += 256) {
    float a = b2[o];
    for (int k = 0; k < FF; ++k) a += h1[k] * w2[k * FF + o];
    h2[o] = a > 0.f ? a : 0.f;
  }
  __syncthreads();
  float p = 0.f;
  for (int k = threadIdx.x; k < FF; k += 256) p += h2[k] * w3[k];
  red[threadIdx.x] = p; __syncthreads();
  for (int s = 128; s > 0; s >>= 1) {
    if (threadIdx.x < s) red[threadIdx.x] += red[threadIdx.x + s];
    __syncthreads();
  }
  if (threadIdx.x == 0) out[r] = red[0] + b3[0];
}

// ---------------- pack fp32 weight [Kreal,Nreal] into WMMA B fragments ----------
// dst flat index = ((kt*NT + nt)*32 + lane)*16 + e
// lane 0..15 -> col = nt*16+lane, K = kt*32 + e (e<8), kt*32 + 16 + (e-8) (e>=8)
// lane 16..31 -> same col, K = kt*32 + 8 + e (e<8), kt*32 + 24 + (e-8) (e>=8)
__global__ void k_pack(const float* __restrict__ w, int Kreal, int Nreal,
                       int KT, int NT, __bf16* __restrict__ dst) {
  int total = KT * NT * 512;
  int f = blockIdx.x * blockDim.x + threadIdx.x;
  if (f >= total) return;
  int e = f & 15;
  int lane = (f >> 4) & 31;
  int t = f >> 9;
  int nt = t % NT;
  int kt = t / NT;
  int col = nt * 16 + (lane & 15);
  int k = kt * 32 + (lane >> 4) * 8 + (e < 8 ? e : 16 + (e - 8));
  float v = (k < Kreal && col < Nreal) ? w[(size_t)k * Nreal + col] : 0.f;
  dst[f] = (__bf16)v;
}

// ---------------- span embedding: one block per candidate ----------------
__global__ void k_span(const float* __restrict__ md, const int* __restrict__ st,
                       const int* __restrict__ en, const float* __restrict__ swe,
                       const float* __restrict__ hs,
                       float* __restrict__ sf32, __bf16* __restrict__ sb16) {
  __shared__ float attn[32];
  int c = blockIdx.x;
  int s = st[c], e = en[c];
  int wid = e - s;                       // in [0, MAXW)
  if (threadIdx.x < 32) {
    int j = threadIdx.x;
    float v = -__builtin_inff();
    if (j <= wid && j < MAXW) {
      int p = s + j; if (p > NW - 1) p = NW - 1;
      v = hs[p];
    }
    float m = v;
    for (int o = 16; o > 0; o >>= 1) { float t = __shfl_xor(m, o, 32); m = m > t ? m : t; }
    float ex = (j <= wid && j < MAXW) ? __expf(v - m) : 0.f;
    float su = ex;
    for (int o = 16; o > 0; o >>= 1) su += __shfl_xor(su, o, 32);
    attn[j] = ex / su;
  }
  __syncthreads();
  float* frow = sf32 + (size_t)c * SD;
  __bf16* brow = sb16 + (size_t)c * KP1;
  for (int h = threadIdx.x; h < HID; h += 256) {
    float sv = md[(size_t)s * HID + h];
    float ev = md[(size_t)e * HID + h];
    float av = 0.f;
    for (int j = 0; j <= wid; ++j) {
      int p = s + j; if (p > NW - 1) p = NW - 1;
      av += attn[j] * md[(size_t)p * HID + h];
    }
    frow[h] = sv;           brow[h] = (__bf16)sv;
    frow[HID + h] = ev;     brow[HID + h] = (__bf16)ev;
    frow[2 * HID + FEAT + h] = av; brow[2 * HID + FEAT + h] = (__bf16)av;
  }
  for (int f = threadIdx.x; f < FEAT; f += 256) {
    float wv = swe[wid * FEAT + f];
    frow[2 * HID + f] = wv; brow[2 * HID + f] = (__bf16)wv;
  }
  for (int k = SD + threadIdx.x; k < KP1; k += 256) brow[k] = (__bf16)0.f;  // K pad
}

// ---------------- bf16 WMMA GEMM ------------------------------------------
// Block = 8 waves, block tile M=256 x N=64. Each wave: 32x64 (2 mt x 4 nt),
// 8 fp32 accumulators, 8 WMMAs per K-step. B K-chunks (4KB) are staged into
// LDS with async global->LDS copies (double buffered), shared by all 8 waves.
__global__ void __launch_bounds__(256)
k_gemm(const __bf16* __restrict__ A, int lda,
       const __bf16* __restrict__ Bp, int KT,
       const float* __restrict__ bias, int Nreal,
       __bf16* __restrict__ outb, float* __restrict__ outf, int ldo) {
  __shared__ alignas(32) __bf16 lbs[2][2048];     // 2 x 4KB double buffer
  const int tid  = threadIdx.x;
  const int wv   = tid >> 5;
  const int lane = tid & 31;
  const int half = lane >> 4;
  const int bm   = (blockIdx.x >> 4) * 256;       // 32 blocks in M
  const int ntb  = (blockIdx.x & 15) * 4;         // 16 blocks in N (4 nt tiles)
  const int row0 = bm + wv * 32 + (lane & 15);
  const __bf16* arow0 = A + (size_t)row0 * lda + half * 8;
  const __bf16* arow1 = arow0 + (size_t)16 * lda;
  v8f acc[2][4] = {};

  // stage B chunk kt (4 fragments = 4KB) into lbs[buf]; 16B per thread
  auto stage = [&](int kt, int buf) {
    const __bf16* gsrc = Bp + ((size_t)kt * NTI + ntb) * 512 + tid * 8;
    __bf16* ldst = &lbs[buf][tid * 8];
#if HAVE_ASYNC
    __builtin_amdgcn_global_load_async_to_lds_b128((v4i*)gsrc, (v4i*)ldst, 0, 0);
#else
    *(v8bf*)ldst = *(const v8bf*)gsrc;
#endif
  };

  stage(0, 0);
  async_wait0();
  __syncthreads();

  for (int kt = 0; kt < KT; ++kt) {
    const int buf = kt & 1;
    if (kt + 1 < KT) stage(kt + 1, buf ^ 1);     // prefetch next chunk

    v16bf a0 = load_afrag(arow0 + kt * 32);
    v16bf a1 = load_afrag(arow1 + kt * 32);
    const __bf16* lb = &lbs[buf][lane * 16];
    v16bf b0 = *(const v16bf*)(lb);
    v16bf b1 = *(const v16bf*)(lb + 512);
    v16bf b2 = *(const v16bf*)(lb + 1024);
    v16bf b3 = *(const v16bf*)(lb + 1536);

    acc[0][0] = __builtin_amdgcn_wmma_f32_16x16x32_bf16(false, a0, false, b0, (short)0, acc[0][0], false, false);
    acc[0][1] = __builtin_amdgcn_wmma_f32_16x16x32_bf16(false, a0, false, b1, (short)0, acc[0][1], false, false);
    acc[0][2] = __builtin_amdgcn_wmma_f32_16x16x32_bf16(false, a0, false, b2, (short)0, acc[0][2], false, false);
    acc[0][3] = __builtin_amdgcn_wmma_f32_16x16x32_bf16(false, a0, false, b3, (short)0, acc[0][3], false, false);
    acc[1][0] = __builtin_amdgcn_wmma_f32_16x16x32_bf16(false, a1, false, b0, (short)0, acc[1][0], false, false);
    acc[1][1] = __builtin_amdgcn_wmma_f32_16x16x32_bf16(false, a1, false, b1, (short)0, acc[1][1], false, false);
    acc[1][2] = __builtin_amdgcn_wmma_f32_16x16x32_bf16(false, a1, false, b2, (short)0, acc[1][2], false, false);
    acc[1][3] = __builtin_amdgcn_wmma_f32_16x16x32_bf16(false, a1, false, b3, (short)0, acc[1][3], false, false);

    async_wait0();      // prefetch for kt+1 complete
    __syncthreads();    // all waves done reading lbs[buf], lbs[buf^1] ready
  }

  for (int ni = 0; ni < 4; ++ni) {
    int col = (ntb + ni) * 16 + (lane & 15);
    float bv = (col < Nreal && bias) ? bias[col] : 0.f;
    for (int mi = 0; mi < 2; ++mi) {
      int rbase = bm + wv * 32 + mi * 16 + half * 8;
      for (int v = 0; v < 8; ++v) {
        float x = acc[mi][ni][v] + bv;
        x = x > 0.f ? x : 0.f;
        if (col >= Nreal) x = 0.f;               // keep pads exactly zero
        if (outb) outb[(size_t)(rbase + v) * ldo + col] = (__bf16)x;
        if (outf) outf[(size_t)(rbase + v) * ldo + col] = x;
      }
    }
  }
}

// ---------------- final score per candidate ----------------
__global__ void k_scores(const float* __restrict__ h2, const float* __restrict__ w3,
                         const float* __restrict__ b3, const float* __restrict__ wsc,
                         const int* __restrict__ st, const int* __restrict__ en,
                         float* __restrict__ out) {
  __shared__ float red[256];
  int c = blockIdx.x;
  float p = 0.f;
  for (int k = threadIdx.x; k < FF; k += 256) p += h2[(size_t)c * NP + k] * w3[k];
  red[threadIdx.x] = p; __syncthreads();
  for (int s = 128; s > 0; s >>= 1) {
    if (threadIdx.x < s) red[threadIdx.x] += red[threadIdx.x + s];
    __syncthreads();
  }
  if (threadIdx.x == 0) out[c] = red[0] + b3[0] + wsc[en[c] - st[c]];
}

// ---------------- single-block bitonic sort of 8192 (score desc, idx asc) ----
__global__ void k_sort(const float* __restrict__ scores, int* __restrict__ sorted_idx) {
  __shared__ unsigned long long key[NC];     // 64 KB
  int tid = threadIdx.x;
  for (int i = tid; i < NC; i += 1024) {
    unsigned u = __float_as_uint(scores[i]);
    u = (u & 0x80000000u) ? ~u : (u | 0x80000000u);   // ascending-order map
    unsigned d = ~u;                                  // descending
    key[i] = ((unsigned long long)d << 32) | (unsigned)i;
  }
  __syncthreads();
  for (unsigned k = 2; k <= NC; k <<= 1) {
    for (unsigned j = k >> 1; j > 0; j >>= 1) {
      for (unsigned i = tid; i < NC; i += 1024) {
        unsigned ixj = i ^ j;
        if (ixj > i) {
          bool up = ((i & k) == 0);
          unsigned long long a = key[i], b = key[ixj];
          if ((a > b) == up) { key[i] = b; key[ixj] = a; }
        }
      }
      __syncthreads();
    }
  }
  for (int i = tid; i < NC; i += 1024) sorted_idx[i] = (int)(key[i] & 0xFFFFFFFFu);
}

// ---------------- greedy crossing-interval NMS + final sort + outputs --------
__global__ void k_nms(const int* __restrict__ sorted_idx,
                      const int* __restrict__ st, const int* __restrict__ en,
                      const float* __restrict__ scores,
                      int* __restrict__ top_idx_ws, float* __restrict__ dout) {
  __shared__ int max_end[NW];
  __shared__ int min_start[NW];
  __shared__ int sel[TOPN + 1];
  __shared__ unsigned long long skey[1024];
  __shared__ int s_count;
  int tid = threadIdx.x;
  for (int i = tid; i < NW; i += 1024) { max_end[i] = -1; min_start[i] = NW; }
  __syncthreads();

  if (tid < 32) {
    int lane = tid;
    int count = 0;
    for (int i = 0; i < NC && count < TOPN; ++i) {
      int idx = sorted_idx[i];
      int s = st[idx], e = en[idx];
      bool dup = (max_end[s] == e);
      bool cr = false;
      if (lane < MAXW) {
        int jr = s + lane;
        if (jr <= e) {
          int jc = jr < NW ? jr : NW - 1;
          int me = max_end[jc], ms = min_start[jc];
          cr = ((jr > s) && (me > e)) || ((jr < e) && (ms < s));
        }
      }
      bool crossed = (__ballot(cr) != 0ull);
      if (!dup && !crossed) {
        if (lane == 0) {
          sel[count] = idx;
          if (e > max_end[s]) max_end[s] = e;
          if (s < min_start[e]) min_start[e] = s;
        }
        __threadfence_block();
        count++;
      }
    }
    if (lane == 0) s_count = count;
  }
  __syncthreads();
  int count = s_count;

  // final sort of selected by (start*NW + end), invalid slots -> INT_MAX
  unsigned key32 = 0x7FFFFFFFu;
  if (tid < count) key32 = (unsigned)(st[sel[tid]] * NW + en[sel[tid]]);
  skey[tid] = ((unsigned long long)key32 << 32) | (unsigned)tid;
  __syncthreads();
  for (unsigned k = 2; k <= 1024; k <<= 1) {
    for (unsigned j = k >> 1; j > 0; j >>= 1) {
      unsigned ixj = tid ^ j;
      if (ixj > (unsigned)tid) {
        bool up = ((tid & k) == 0);
        unsigned long long a = skey[tid], b = skey[ixj];
        if ((a > b) == up) { skey[tid] = b; skey[ixj] = a; }
      }
      __syncthreads();
    }
  }

  if (tid < TOPN) {
    int slot = (tid < count) ? (int)(skey[tid] & 0xFFFFFFFFu)
                             : (int)(skey[0] & 0xFFFFFFFFu);
    int v = sel[slot];
    top_idx_ws[tid] = v;
    dout[O_TIDX + tid]   = (float)v;
    dout[O_TSTART + tid] = (float)st[v];
    dout[O_TEND + tid]   = (float)en[v];
    dout[O_TSCORE + tid] = scores[v];
  }
}

// ---------------- gather top span embeddings ----------------
__global__ void k_gather(const int* __restrict__ top_idx_ws,
                         const float* __restrict__ sf32, float* __restrict__ dout) {
  int b = blockIdx.x;
  const float* src = sf32 + (size_t)top_idx_ws[b] * SD;
  float* dst = dout + O_TEMB + (size_t)b * SD;
  for (int h = threadIdx.x; h < SD; h += 256) dst[h] = src[h];
}

// ---------------- host launch ----------------
extern "C" void kernel_launch(void* const* d_in, const int* in_sizes, int n_in,
                              void* d_out, int out_size, void* d_ws, size_t ws_size,
                              hipStream_t stream) {
  const float* mention_doc = (const float*)d_in[0];
  const float* span_width_emb = (const float*)d_in[1];
  const float* head_w = (const float*)d_in[2];
  const float* head_b = (const float*)d_in[3];
  const float* m_w1 = (const float*)d_in[4];
  const float* m_b1 = (const float*)d_in[5];
  const float* m_w2 = (const float*)d_in[6];
  const float* m_b2 = (const float*)d_in[7];
  const float* m_w3 = (const float*)d_in[8];
  const float* m_b3 = (const float*)d_in[9];
  const float* width_prior_emb = (const float*)d_in[10];
  const float* w_w1 = (const float*)d_in[11];
  const float* w_b1 = (const float*)d_in[12];
  const float* w_w2 = (const float*)d_in[13];
  const float* w_b2 = (const float*)d_in[14];
  const float* w_w3 = (const float*)d_in[15];
  const float* w_b3 = (const float*)d_in[16];
  const int* cand_starts = (const int*)d_in[17];
  const int* cand_ends = (const int*)d_in[18];
  float* out = (float*)d_out;

  // workspace carve-up (all region sizes multiples of 256B)
  char* w = (char*)d_ws;
  float* head_scores = (float*)w;             w += NW * 4;                     // 8KB
  float* width_scores = (float*)w;            w += 64 * 4;                     // 256B
  int*   sorted_idx = (int*)w;                w += NC * 4;                     // 32KB
  int*   top_idx_ws = (int*)w;                w += 1024 * 4;                   // 4KB
  float* span_f32 = (float*)w;                w += (size_t)NC * SD * 4;        // 76.2MB
  __bf16* span_b16 = (__bf16*)w;              w += (size_t)NC * KP1 * 2;       // 38.3MB
  __bf16* w1p = (__bf16*)w;                   w += (size_t)KT1 * NTI * 512 * 2;// 4.8MB
  __bf16* w2p = (__bf16*)w;                   w += (size_t)KT2 * NTI * 512 * 2;// 2MB
  __bf16* h1b = (__bf16*)w;                   w += (size_t)NC * NP * 2;        // 16.8MB
  float* h2f = (float*)w;                     w += (size_t)NC * NP * 4;        // 33.6MB

  // 1) head scores [2048]
  k_head<<<NW, 256, 0, stream>>>(mention_doc, head_w, head_b, head_scores);
  // 2) width prior FFNN [30]
  k_width<<<MAXW, 256, 0, stream>>>(width_prior_emb, w_w1, w_b1, w_w2, w_b2,
                                    w_w3, w_b3, width_scores);
  // 3) pack weights into WMMA B-fragment layout (bf16)
  {
    int t1 = KT1 * NTI * 512;
    k_pack<<<(t1 + 255) / 256, 256, 0, stream>>>(m_w1, SD, FF, KT1, NTI, w1p);
    int t2 = KT2 * NTI * 512;
    k_pack<<<(t2 + 255) / 256, 256, 0, stream>>>(m_w2, FF, FF, KT2, NTI, w2p);
  }
  // 4) span embeddings (fp32 for output gather + bf16 K-padded for GEMM A)
  k_span<<<NC, 256, 0, stream>>>(mention_doc, cand_starts, cand_ends,
                                 span_width_emb, head_scores, span_f32, span_b16);
  // 5) GEMM1: [8192,2336] x [2336,1024] -> relu -> bf16 hidden1
  k_gemm<<<512, 256, 0, stream>>>(span_b16, KP1, w1p, KT1,
                                  m_b1, FF, h1b, (float*)nullptr, NP);
  // 6) GEMM2: [8192,1024] x [1024,1024] -> relu -> f32 hidden2
  k_gemm<<<512, 256, 0, stream>>>(h1b, NP, w2p, KT2,
                                  m_b2, FF, (__bf16*)nullptr, h2f, NP);
  // 7) mention scores + width prior -> d_out[0:8192]
  k_scores<<<NC, 256, 0, stream>>>(h2f, m_w3, m_b3, width_scores,
                                   cand_starts, cand_ends, out + O_SCORES);
  // 8) sort candidates by score (desc, stable)
  k_sort<<<1, 1024, 0, stream>>>(out + O_SCORES, sorted_idx);
  // 9) greedy NMS + final (start,end) sort + scalar outputs
  k_nms<<<1, 1024, 0, stream>>>(sorted_idx, cand_starts, cand_ends,
                                out + O_SCORES, top_idx_ws, out);
  // 10) gather top span embeddings
  k_gather<<<TOPN, 256, 0, stream>>>(top_idx_ws, span_f32, out);
}
